// Qwen3_17463337025856
// MI455X (gfx1250) — matrix operand order repaired and checked by reading, not compiled, not run on hardware
//
#include <hip/hip_runtime.h>
#include <hip/hip_bf16.h>
#include <math.h>

// ---- model dims ----
#define SEQ   2048
#define DM    2048
#define HD    128
#define QH    16
#define KVH   8
#define NL    2
#define FFN   6144
#define VOCAB 32000
#define THETA 1000000.0f
#define EPSV  1e-6f
#define CHUNK_V 6400   // 32000 = 5 * 6400; chunk (52MB) stays resident in 192MB L2

#define KT    64       // GEMM K-stage
#define KTP   72       // padded LDS row (16B-aligned pad, breaks bank pattern)

typedef __attribute__((ext_vector_type(16))) __bf16 bf16x16;
typedef __attribute__((ext_vector_type(8)))  float  f32x8;

union AFragU { bf16x16 v; uint4 q[2]; __bf16 e[16]; };

static __device__ inline f32x8 zero8() {
  f32x8 z; for (int i = 0; i < 8; ++i) z[i] = 0.0f; return z;
}

// Generic pointers into LDS carry the LDS byte offset in their low 32 bits
// (flat aperture, ISA 10.2).
static __device__ inline unsigned lds_off(const void* p) {
  return (unsigned)(unsigned long long)p;
}

// ASYNCcnt-tracked DMA of 16B global -> LDS (no VGPR round trip).
static __device__ inline void async_b128(unsigned loff, const void* g) {
  asm volatile("global_load_async_to_lds_b128 %0, %1, off"
               :: "v"(loff), "v"(g) : "memory");
}

// A-fragment (16x32 bf16, MxK): lanes 0-15 -> rows M=0..15 holding K=0..7,16..23;
// lanes 16-31 -> K=8..15,24..31 (ISA 05_wmma "16-bit A-Matrix 16x32" layout).
static __device__ inline bf16x16 load_frag_a(const __bf16* A, int lda,
                                             int m0, int k0, int lane) {
  const int r  = lane & 15;
  const int hi = lane >> 4;
  const __bf16* p = A + (size_t)(m0 + r) * lda + k0 + hi * 8;
  AFragU f;
  f.q[0] = *reinterpret_cast<const uint4*>(p);       // K = base .. base+7
  f.q[1] = *reinterpret_cast<const uint4*>(p + 16);  // K = base+16 .. base+23
  return f.v;
}

// B-fragment (32x16 bf16, KxN): lane n=lane&15 is column N; lanes 0-15 hold K=0..7,16..23,
// lanes 16-31 hold K=8..15,24..31 (same order as A). B stored row-major N x K.
static __device__ inline bf16x16 load_frag_b(const __bf16* B, int ldb,
                                             int n0, int k0, int lane) {
  const int n  = lane & 15;
  const int hi = lane >> 4;
  const __bf16* p = B + (size_t)(n0 + n) * ldb + k0 + hi * 8;
  AFragU f;
  f.q[0] = *reinterpret_cast<const uint4*>(p);
  f.q[1] = *reinterpret_cast<const uint4*>(p + 16);
  return f.v;
}

static __device__ inline f32x8 wmma_bf16(bf16x16 a, bf16x16 b, f32x8 c) {
  return __builtin_amdgcn_wmma_f32_16x16x32_bf16(false, a, false, b, (short)0, c,
                                                 false, false);
}

// ---------------- utility kernels ----------------

__global__ void k_f32_to_bf16(const float* __restrict__ in, __bf16* __restrict__ out,
                              size_t n) {
  size_t i = (size_t)blockIdx.x * blockDim.x + threadIdx.x;
  size_t stride = (size_t)gridDim.x * blockDim.x;
  for (; i < n; i += stride) out[i] = (__bf16)in[i];
}

__global__ void k_embed(const int* __restrict__ tok, const float* __restrict__ emb,
                        float* __restrict__ x) {
  int i = blockIdx.x * blockDim.x + threadIdx.x;
  if (i < SEQ * DM) {
    int s = i / DM, d = i % DM;
    x[i] = emb[(size_t)tok[s] * DM + d];
  }
}

// RMSNorm row-wise (fp32 in, bf16 out), one block per row.
__global__ __launch_bounds__(256) void k_rmsnorm(const float* __restrict__ x,
                                                 const float* __restrict__ w,
                                                 __bf16* __restrict__ out, int ncols) {
  const int row = blockIdx.x;
  const float* xr = x + (size_t)row * ncols;
  float ss = 0.0f;
  for (int i = threadIdx.x; i < ncols; i += 256) { float v = xr[i]; ss += v * v; }
  for (int m = 16; m >= 1; m >>= 1) ss += __shfl_xor(ss, m, 32);
  __shared__ float part[8];
  if ((threadIdx.x & 31) == 0) part[threadIdx.x >> 5] = ss;
  __syncthreads();
  float tot = 0.0f;
  for (int i = 0; i < 8; ++i) tot += part[i];
  const float rn = rsqrtf(tot / (float)ncols + EPSV);
  for (int i = threadIdx.x; i < ncols; i += 256)
    out[(size_t)row * ncols + i] = (__bf16)(xr[i] * rn * w[i]);
}

// ---------------- WMMA GEMM: C[M,N] (+)= A[M,K] * B[N,K]^T ----------------
// Block = 256 threads = 8 waves (2 M x 4 N); block tile 64(M) x 128(N).
// Operand tiles are DMA'd to LDS with global_load_async_to_lds_b128 (ASYNCcnt),
// double-buffered (K-stage 64), fragments served from LDS via ds_load_b128.
__global__ __launch_bounds__(256) void k_gemm_bf16_nt(const __bf16* __restrict__ A,
                                                      const __bf16* __restrict__ B,
                                                      float* __restrict__ C,
                                                      int M, int N, int K, int accum) {
  __shared__ __bf16 lA[2][64][KTP];    // 18 KB
  __shared__ __bf16 lB[2][128][KTP];   // 36 KB

  const int tid  = threadIdx.x;
  const int lane = tid & 31;
  const int wave = tid >> 5;
  const int mw = (wave & 1) * 32;         // wave row offset within block tile
  const int nw = (wave >> 1) * 32;        // wave col offset within block tile
  const int m0b = blockIdx.y * 64;
  const int n0b = blockIdx.x * 128;

  // Per-stage async DMA: A 64x64 (2 x b128/thread) + B 128x64 (4 x b128/thread).
  const int arow = tid >> 3;              // 0..31 (A: +0/+32), B uses 4 strides
  const int akc  = (tid & 7) * 8;
  const __bf16* gA = A + (size_t)(m0b + arow) * K + akc;
  const __bf16* gB = B + (size_t)(n0b + arow) * K + akc;

  f32x8 acc[2][2];
  for (int i = 0; i < 2; ++i)
    for (int j = 0; j < 2; ++j) acc[i][j] = zero8();

  const int nt = K / KT;

#define STAGE(kt_)                                                              \
  do {                                                                          \
    const int buf_ = (kt_) & 1;                                                 \
    const size_t k0_ = (size_t)(kt_) * KT;                                      \
    async_b128(lds_off(&lA[buf_][arow][akc]),      gA + k0_);                   \
    async_b128(lds_off(&lA[buf_][arow + 32][akc]), gA + k0_ + (size_t)32 * K);  \
    async_b128(lds_off(&lB[buf_][arow][akc]),      gB + k0_);                   \
    async_b128(lds_off(&lB[buf_][arow + 32][akc]), gB + k0_ + (size_t)32 * K);  \
    async_b128(lds_off(&lB[buf_][arow + 64][akc]), gB + k0_ + (size_t)64 * K);  \
    async_b128(lds_off(&lB[buf_][arow + 96][akc]), gB + k0_ + (size_t)96 * K);  \
  } while (0)

  STAGE(0);
  for (int kt = 0; kt < nt; ++kt) {
    if (kt + 1 < nt) {
      STAGE(kt + 1);  // buffers alternate; prior trailing barrier makes this safe
      asm volatile("s_wait_asynccnt 0x6" ::: "memory");  // stage kt done, kt+1 in flight
    } else {
      asm volatile("s_wait_asynccnt 0x0" ::: "memory");
    }
    __syncthreads();
    const __bf16* tA = &lA[kt & 1][0][0];
    const __bf16* tB = &lB[kt & 1][0][0];
#pragma unroll
    for (int kk = 0; kk < KT; kk += 32) {
      bf16x16 a0 = load_frag_a(tA, KTP, mw,      kk, lane);
      bf16x16 a1 = load_frag_a(tA, KTP, mw + 16, kk, lane);
      bf16x16 b0 = load_frag_b(tB, KTP, nw,      kk, lane);
      bf16x16 b1 = load_frag_b(tB, KTP, nw + 16, kk, lane);
      acc[0][0] = wmma_bf16(a0, b0, acc[0][0]);
      acc[0][1] = wmma_bf16(a0, b1, acc[0][1]);
      acc[1][0] = wmma_bf16(a1, b0, acc[1][0]);
      acc[1][1] = wmma_bf16(a1, b1, acc[1][1]);
    }
    __syncthreads();
  }
#undef STAGE

  // C/D layout: VGPR r, lanes 0-15 -> row r, col=lane; lanes 16-31 -> row r+8.
  const int chalf = lane >> 4;
  const int ccol  = lane & 15;
  for (int mi = 0; mi < 2; ++mi)
    for (int ni = 0; ni < 2; ++ni)
      for (int r = 0; r < 8; ++r) {
        const int row = m0b + mw + mi * 16 + r + chalf * 8;
        const int col = n0b + nw + ni * 16 + ccol;
        const size_t idx = (size_t)row * N + col;
        const float v = acc[mi][ni][r];
        C[idx] = accum ? (C[idx] + v) : v;
      }
}

// ---------------- qk-norm + RoPE + V transpose ----------------
// grid (SEQ, QH+2*KVH), block 128 (= HD). Heads 0..15 q, 16..23 k, 24..31 v.
__global__ __launch_bounds__(128) void k_qkv_prep(const float* __restrict__ qkv,
                                                  const float* __restrict__ qknorm_l,
                                                  __bf16* __restrict__ Qb,
                                                  __bf16* __restrict__ Kb,
                                                  __bf16* __restrict__ Vt) {
  const int s = blockIdx.x;
  const int hh = blockIdx.y;
  const int d = threadIdx.x;
  const float* src = qkv + ((size_t)s * (QH + 2 * KVH) + hh) * HD;
  const float v = src[d];
  if (hh < QH + KVH) {
    float ss = v * v;
    for (int m = 16; m >= 1; m >>= 1) ss += __shfl_xor(ss, m, 32);
    __shared__ float part[4];
    if ((threadIdx.x & 31) == 0) part[threadIdx.x >> 5] = ss;
    __syncthreads();
    const float tot = part[0] + part[1] + part[2] + part[3];
    const float rn = rsqrtf(tot / (float)HD + EPSV);
    const float* wn = qknorm_l + (hh < QH ? 0 : HD);
    const float xn = v * rn * wn[d];
    const int dp = (d < 64) ? d + 64 : d - 64;
    const float xnp = src[dp] * rn * wn[dp];
    const int fi = d & 63;
    const float inv = exp2f(-(float)fi * (log2f(THETA) / 64.0f));
    const float ang = (float)s * inv;
    const float c = cosf(ang), sn = sinf(ang);
    const float rot = (d < 64) ? -xnp : xnp;
    const float o = xn * c + rot * sn;
    if (hh < QH) Qb[((size_t)hh * SEQ + s) * HD + d] = (__bf16)o;
    else         Kb[((size_t)(hh - QH) * SEQ + s) * HD + d] = (__bf16)o;
  } else {
    const int h = hh - QH - KVH;
    Vt[((size_t)h * HD + d) * SEQ + s] = (__bf16)v;  // V^T so P*V is also NT GEMM
  }
}

// ---------------- flash attention (causal, GQA G=2) ----------------
// grid (SEQ/64, QH), block 128 = 4 waves; each wave owns 16 query rows x HD=128.
__global__ __launch_bounds__(128) void k_attn(const __bf16* __restrict__ Qb,
                                              const __bf16* __restrict__ Kb,
                                              const __bf16* __restrict__ Vt,
                                              __bf16* __restrict__ Out) {
  const int lane = threadIdx.x & 31;
  const int wave = threadIdx.x >> 5;
  const int h  = blockIdx.y;
  const int kv = h >> 1;  // G = QH/KVH = 2
  const int qbase = blockIdx.x * 64 + wave * 16;
  const float scale = 0.088388347648f;  // 1/sqrt(128)

  const __bf16* Q = Qb + (size_t)h  * SEQ * HD;
  const __bf16* K = Kb + (size_t)kv * SEQ * HD;
  const __bf16* V = Vt + (size_t)kv * HD * SEQ;

  bf16x16 qa[4];
  for (int kc = 0; kc < 4; ++kc) qa[kc] = load_frag_a(Q, HD, qbase, kc * 32, lane);

  f32x8 oacc[8];
  for (int t = 0; t < 8; ++t) oacc[t] = zero8();
  float m[8], l[8];
  for (int r = 0; r < 8; ++r) { m[r] = -1e30f; l[r] = 0.0f; }

  __shared__ float plds[4][16][32];

  const int jend  = blockIdx.x * 64 + 64;  // uniform over block (causal upper bound)
  const int chalf = lane >> 4;
  const int ccol  = lane & 15;

  for (int j0 = 0; j0 < jend; j0 += 32) {
    if (j0 + 32 < jend) {  // next K-tile prefetch -> global_prefetch_b8
      __builtin_prefetch(K + (size_t)(j0 + 32 + ccol) * HD, 0, 1);
    }
    // S tile (16x32) = Q(16x128) * K(j0..j0+31)^T via 4x2 WMMA
    f32x8 s0 = zero8(), s1 = zero8();
    for (int kc = 0; kc < 4; ++kc) {
      bf16x16 kb0 = load_frag_b(K, HD, j0,      kc * 32, lane);
      bf16x16 kb1 = load_frag_b(K, HD, j0 + 16, kc * 32, lane);
      s0 = wmma_bf16(qa[kc], kb0, s0);
      s1 = wmma_bf16(qa[kc], kb1, s1);
    }
    __syncthreads();  // previous iteration finished reading plds
    for (int r = 0; r < 8; ++r) {
      const int row = qbase + r + chalf * 8;
      float v0 = s0[r] * scale; if (j0 + ccol > row)      v0 = -1e30f;
      float v1 = s1[r] * scale; if (j0 + 16 + ccol > row) v1 = -1e30f;
      float mr = fmaxf(v0, v1);
      for (int msk = 8; msk >= 1; msk >>= 1) mr = fmaxf(mr, __shfl_xor(mr, msk, 32));
      const float newm = fmaxf(m[r], mr);
      const float alpha = __expf(m[r] - newm);
      const float p0 = __expf(v0 - newm);
      const float p1 = __expf(v1 - newm);
      float rs = p0 + p1;
      for (int msk = 8; msk >= 1; msk >>= 1) rs += __shfl_xor(rs, msk, 32);
      l[r] = l[r] * alpha + rs;
      m[r] = newm;
      for (int t = 0; t < 8; ++t) oacc[t][r] *= alpha;
      plds[wave][r + chalf * 8][ccol]      = p0;   // C-layout -> LDS
      plds[wave][r + chalf * 8][16 + ccol] = p1;
    }
    __syncthreads();
    // P (16x32) re-read in A-fragment layout, converted to bf16
    AFragU pf;
    {
      const int rr = lane & 15;
      const int cb = (lane >> 4) * 8;
      for (int i = 0; i < 8; ++i) {
        pf.e[i]     = (__bf16)plds[wave][rr][cb + i];
        pf.e[8 + i] = (__bf16)plds[wave][rr][cb + 16 + i];
      }
    }
    // O += P * V : 8 output d-tiles, one WMMA each (K = 32 keys)
    for (int t = 0; t < 8; ++t) {
      bf16x16 vb = load_frag_b(V, SEQ, t * 16, j0, lane);
      oacc[t] = wmma_bf16(pf.v, vb, oacc[t]);
    }
  }

  for (int t = 0; t < 8; ++t)
    for (int r = 0; r < 8; ++r) {
      const int row = qbase + r + chalf * 8;
      const int col = t * 16 + ccol;
      Out[(size_t)row * (QH * HD) + h * HD + col] = (__bf16)(oacc[t][r] / l[r]);
    }
}

// ---------------- SiLU(gate) * up ----------------
__global__ void k_silu_mul(const float* __restrict__ gu, __bf16* __restrict__ hbuf) {
  size_t i = (size_t)blockIdx.x * blockDim.x + threadIdx.x;
  size_t stride = (size_t)gridDim.x * blockDim.x;
  for (; i < (size_t)SEQ * FFN; i += stride) {
    const int s = (int)(i / FFN), f = (int)(i % FFN);
    const float g = gu[(size_t)s * 2 * FFN + f];
    const float u = gu[(size_t)s * 2 * FFN + FFN + f];
    hbuf[i] = (__bf16)(g / (1.0f + __expf(-g)) * u);
  }
}

// ---------------- streaming vocab log-sum-exp loss ----------------
__global__ void k_loss_init(float* stats) {
  int s = blockIdx.x * blockDim.x + threadIdx.x;
  if (s < SEQ) { stats[s * 3] = -1e30f; stats[s * 3 + 1] = 0.0f; stats[s * 3 + 2] = 0.0f; }
}

__global__ __launch_bounds__(256) void k_loss_chunk(const float* __restrict__ logits,
                                                    const int* __restrict__ tgt,
                                                    float* __restrict__ stats,
                                                    int base, int n) {
  const int s = blockIdx.x;
  const float* row = logits + (size_t)s * n;
  float mx = -1e30f;
  for (int i = threadIdx.x; i < n; i += 256) mx = fmaxf(mx, row[i]);
  for (int msk = 16; msk >= 1; msk >>= 1) mx = fmaxf(mx, __shfl_xor(mx, msk, 32));
  __shared__ float part[8];
  if ((threadIdx.x & 31) == 0) part[threadIdx.x >> 5] = mx;
  __syncthreads();
  for (int i = 0; i < 8; ++i) mx = fmaxf(mx, part[i]);
  const float oldm = stats[s * 3];
  const float newm = fmaxf(oldm, mx);
  float se = 0.0f;
  for (int i = threadIdx.x; i < n; i += 256) se += __expf(row[i] - newm);
  for (int msk = 16; msk >= 1; msk >>= 1) se += __shfl_xor(se, msk, 32);
  __syncthreads();
  if ((threadIdx.x & 31) == 0) part[threadIdx.x >> 5] = se;
  __syncthreads();
  if (threadIdx.x == 0) {
    float tot = 0.0f;
    for (int i = 0; i < 8; ++i) tot += part[i];
    stats[s * 3 + 1] = stats[s * 3 + 1] * __expf(oldm - newm) + tot;
    stats[s * 3] = newm;
    const int t = tgt[s] - base;
    if (t >= 0 && t < n) stats[s * 3 + 2] = row[t];
  }
}

__global__ __launch_bounds__(256) void k_loss_final(const float* __restrict__ stats,
                                                    float* __restrict__ out) {
  float acc = 0.0f;
  for (int s = threadIdx.x; s < SEQ; s += 256)
    acc += -(stats[s * 3 + 2] - stats[s * 3] - logf(stats[s * 3 + 1]));
  for (int msk = 16; msk >= 1; msk >>= 1) acc += __shfl_xor(acc, msk, 32);
  __shared__ float part[8];
  if ((threadIdx.x & 31) == 0) part[threadIdx.x >> 5] = acc;
  __syncthreads();
  if (threadIdx.x == 0) {
    float tot = 0.0f;
    for (int i = 0; i < 8; ++i) tot += part[i];
    out[0] = tot / (float)SEQ;
  }
}

// ---------------- host driver ----------------
static inline size_t alignup(size_t x) { return (x + 255) & ~(size_t)255; }

extern "C" void kernel_launch(void* const* d_in, const int* in_sizes, int n_in,
                              void* d_out, int out_size, void* d_ws, size_t ws_size,
                              hipStream_t stream) {
  const int*   tok  = (const int*)d_in[0];
  const int*   tgt  = (const int*)d_in[1];
  const float* emb  = (const float*)d_in[2];
  const float* lm   = (const float*)d_in[3];
  const float* rmsA = (const float*)d_in[4];
  const float* rmsF = (const float*)d_in[5];
  const float* qkn  = (const float*)d_in[6];
  const float* wqkv = (const float*)d_in[7];
  const float* wo   = (const float*)d_in[8];
  const float* wgu  = (const float*)d_in[9];
  const float* wdn  = (const float*)d_in[10];

  char* p = (char*)d_ws;
  auto take = [&](size_t bytes) -> char* { char* r = p; p += alignup(bytes); return r; };

  const size_t n_wqkv = (size_t)NL * (QH + 2 * KVH) * HD * DM;
  const size_t n_wo   = (size_t)NL * DM * (QH * HD);
  const size_t n_wgu  = (size_t)NL * 2 * FFN * DM;
  const size_t n_wdn  = (size_t)NL * DM * FFN;
  const size_t n_wlm  = (size_t)VOCAB * DM;

  __bf16* bqkv = (__bf16*)take(2 * n_wqkv);
  __bf16* bwo  = (__bf16*)take(2 * n_wo);
  __bf16* bwgu = (__bf16*)take(2 * n_wgu);
  __bf16* bwdn = (__bf16*)take(2 * n_wdn);
  __bf16* bwlm = (__bf16*)take(2 * n_wlm);

  float*  x       = (float*)take(4ull * SEQ * DM);
  __bf16* xb      = (__bf16*)take(2ull * SEQ * DM);
  float*  qkvbuf  = (float*)take(4ull * SEQ * (QH + 2 * KVH) * HD);
  __bf16* Qb      = (__bf16*)take(2ull * QH * SEQ * HD);
  __bf16* Kb      = (__bf16*)take(2ull * KVH * SEQ * HD);
  __bf16* Vt      = (__bf16*)take(2ull * KVH * HD * SEQ);
  __bf16* attnout = (__bf16*)take(2ull * SEQ * QH * HD);
  float*  gubuf   = (float*)take(4ull * SEQ * 2 * FFN);
  __bf16* hbuf    = (__bf16*)take(2ull * SEQ * FFN);
  float*  lchunk  = (float*)take(4ull * SEQ * CHUNK_V);
  float*  stats   = (float*)take(4ull * SEQ * 3);

  auto conv = [&](const float* src, __bf16* dst, size_t n) {
    k_f32_to_bf16<<<2048, 256, 0, stream>>>(src, dst, n);
  };
  auto gemm = [&](const __bf16* A, const __bf16* B, float* C, int M, int N, int K,
                  int accum) {
    dim3 g(N / 128, M / 64);
    k_gemm_bf16_nt<<<g, 256, 0, stream>>>(A, B, C, M, N, K, accum);
  };

  // one-time fp32 -> bf16 weight conversion (compute path is bf16 WMMA)
  conv(wqkv, bqkv, n_wqkv);
  conv(wo,   bwo,  n_wo);
  conv(wgu,  bwgu, n_wgu);
  conv(wdn,  bwdn, n_wdn);
  conv(lm,   bwlm, n_wlm);

  k_embed<<<(SEQ * DM + 255) / 256, 256, 0, stream>>>(tok, emb, x);

  for (int l = 0; l < NL; ++l) {
    k_rmsnorm<<<SEQ, 256, 0, stream>>>(x, rmsA + (size_t)l * DM, xb, DM);
    gemm(xb, bqkv + (size_t)l * (QH + 2 * KVH) * HD * DM, qkvbuf,
         SEQ, (QH + 2 * KVH) * HD, DM, 0);
    k_qkv_prep<<<dim3(SEQ, QH + 2 * KVH), 128, 0, stream>>>(
        qkvbuf, qkn + (size_t)l * 2 * HD, Qb, Kb, Vt);
    k_attn<<<dim3(SEQ / 64, QH), 128, 0, stream>>>(Qb, Kb, Vt, attnout);
    gemm(attnout, bwo + (size_t)l * DM * (QH * HD), x, SEQ, DM, QH * HD, 1);
    k_rmsnorm<<<SEQ, 256, 0, stream>>>(x, rmsF + (size_t)l * DM, xb, DM);
    gemm(xb, bwgu + (size_t)l * 2 * FFN * DM, gubuf, SEQ, 2 * FFN, DM, 0);
    k_silu_mul<<<4096, 256, 0, stream>>>(gubuf, hbuf);
    gemm(hbuf, bwdn + (size_t)l * DM * FFN, x, SEQ, DM, FFN, 1);
  }

  k_rmsnorm<<<SEQ, 256, 0, stream>>>(x, rmsA + (size_t)NL * DM, xb, DM);

  k_loss_init<<<(SEQ + 255) / 256, 256, 0, stream>>>(stats);
  for (int c = 0; c < VOCAB / CHUNK_V; ++c) {
    gemm(xb, bwlm + (size_t)c * CHUNK_V * DM, lchunk, SEQ, CHUNK_V, DM, 0);
    k_loss_chunk<<<SEQ, 256, 0, stream>>>(lchunk, tgt, stats, c * CHUNK_V, CHUNK_V);
  }
  k_loss_final<<<1, 256, 0, stream>>>(stats, (float*)d_out);
}
